// IntraClassEncoder_8383776162282
// MI455X (gfx1250) — matrix operand order repaired and checked by reading, not compiled, not run on hardware
//
#include <hip/hip_runtime.h>
#include <hip/hip_bf16.h>
#include <math.h>

typedef __attribute__((ext_vector_type(16))) _Float16 v16h;
typedef __attribute__((ext_vector_type(8)))  float    v8f;

#define NS   4
#define NM   1024
#define NP   20
#define NTL  64
#define NA   64
#define NTI  11
#define DMA  32
#define DAA  68
#define DTLA 24
#define HD   256
#define NH   4
#define KNN  36
#define FFD  1024
#define NTOT (NM + NTL + NA)
#define NEGB (-1e9f)

// ---------------------------------------------------------------------------
// WMMA helpers (CDNA5 v_wmma_f32_16x16x32_f16, wave32)
// ---------------------------------------------------------------------------
__device__ __forceinline__ v8f wmma_f16(v16h a, v16h b, v8f c) {
  return __builtin_amdgcn_wmma_f32_16x16x32_f16(false, a, false, b, (short)0, c,
                                                false, false);
}

// A fragment (16x32 f16) from f16 LDS, row-major, fully padded (no guards).
__device__ __forceinline__ v16h a_frag_h(const _Float16* A, int lda, int r0, int k0,
                                         int lane) {
  int r = r0 + (lane & 15);
  int kh = (lane >> 4) * 8;
  const _Float16* base = A + r * lda + k0 + kh;
  v16h a;
#pragma unroll
  for (int j = 0; j < 8; ++j) a[j] = base[j];
#pragma unroll
  for (int j = 8; j < 16; ++j) a[j] = base[16 + (j - 8)];
  return a;
}

// B fragment from pre-packed weights: contiguous 32 bytes per lane.
// Packed layout: [ct][kt][lane][16 halves]
__device__ __forceinline__ v16h b_frag_p(const _Float16* __restrict__ Wp, int ktc,
                                         int kt, int ct, int lane) {
  return *(const v16h*)(Wp + (((size_t)ct * ktc + kt) << 9) + lane * 16);
}

// C/D layout: VGPR i -> row r0 + (lane>=16 ? 8:0) + i, col c0 + (lane&15)
__device__ __forceinline__ void store_h(_Float16* D, int ld, int r0, int c0, v8f acc,
                                        const float* bias, bool relu, int lane) {
  int r = r0 + (lane >> 4) * 8;
  int c = c0 + (lane & 15);
  float bv = bias ? bias[c] : 0.0f;
#pragma unroll
  for (int i = 0; i < 8; ++i) {
    float v = acc[i] + bv;
    if (relu) v = fmaxf(v, 0.0f);
    D[(r + i) * ld + c] = (_Float16)v;
  }
}

// CDNA5 async global->LDS copy (ASYNCcnt-tracked, 16B per lane)
__device__ __forceinline__ void async_stage16(const _Float16* src, _Float16* dstLds) {
  unsigned d = (unsigned)(size_t)dstLds;  // LDS offset = addr[31:0]
  unsigned long long g = (unsigned long long)src;
  asm volatile("global_load_async_to_lds_b128 %0, %1, off"
               :: "v"(d), "v"(g)
               : "memory");
}
__device__ __forceinline__ void wait_async0() {
  asm volatile("s_wait_asynccnt 0" ::: "memory");
}

// ---------------------------------------------------------------------------
// Weight packing: W[K,N] f32 -> fragment-ordered f16, K padded to Kpad
// ---------------------------------------------------------------------------
__global__ void pack_b(const float* __restrict__ W, _Float16* __restrict__ Wp,
                       int N, int K, int Kpad) {
  int idx = blockIdx.x * 256 + threadIdx.x;
  int ktc = Kpad >> 5;
  int total = (N >> 4) * ktc * 32;
  if (idx >= total) return;
  int lane = idx & 31;
  int kt = (idx >> 5) % ktc;
  int ct = (idx >> 5) / ktc;
  int c = ct * 16 + (lane & 15);
  int kb = kt * 32 + (lane >> 4) * 8;
  _Float16* dst = Wp + (size_t)idx * 16;
#pragma unroll
  for (int j = 0; j < 16; ++j) {
    int k = kb + (j < 8 ? j : 8 + j);
    dst[j] = (_Float16)((k < K) ? W[(size_t)k * N + c] : 0.0f);
  }
}

// ---------------------------------------------------------------------------
// Generic GEMM: out = act(Ah[M,K](f16) @ W[K,N] + bias (+res f32))
// M%64==0, N%128==0, K%32==0. grid=(M/64, N/128), block=256 (8 waves).
// A tiles (64x32 f16) double-buffered into LDS via async copy; one B fragment
// per K-step feeds 4 wmma. Optional f32 and f16 outputs.
// ---------------------------------------------------------------------------
__global__ void gemm_ws(const _Float16* __restrict__ Ah, const _Float16* __restrict__ Wp,
                        const float* __restrict__ bias, const float* __restrict__ res,
                        float* __restrict__ out, _Float16* __restrict__ outh,
                        int N, int K, int act) {
  __shared__ __align__(16) _Float16 As[2][64 * 32];
  int lane = threadIdx.x & 31, wave = threadIdx.x >> 5;
  int r0 = blockIdx.x * 64;
  int c0 = blockIdx.y * 128 + wave * 16;
  int ct = c0 >> 4;
  int ktc = K >> 5;
  int sr = threadIdx.x >> 2;        // 0..63 staging row
  int sc = (threadIdx.x & 3) * 8;   // 0,8,16,24 staging col base (halves)
  const _Float16* srcb = Ah + (size_t)(r0 + sr) * K + sc;
  _Float16* d0 = &As[0][sr * 32 + sc];
  _Float16* d1 = &As[1][sr * 32 + sc];
  v8f acc[4] = {};
  async_stage16(srcb, d0);
  wait_async0();
  __syncthreads();
  for (int kt = 0; kt < ktc; ++kt) {
    int cur = kt & 1;
    if (kt + 1 < ktc)
      async_stage16(srcb + ((kt + 1) << 5), cur ? d0 : d1);
    v16h b = b_frag_p(Wp, ktc, kt, ct, lane);
    const _Float16* Ab = As[cur];
#pragma unroll
    for (int mt = 0; mt < 4; ++mt) {
      v16h a = a_frag_h(Ab, 32, mt * 16, 0, lane);
      acc[mt] = wmma_f16(a, b, acc[mt]);
    }
    wait_async0();
    __syncthreads();
  }
  int c = c0 + (lane & 15);
  float bv = bias ? bias[c] : 0.0f;
#pragma unroll
  for (int mt = 0; mt < 4; ++mt) {
    int r = r0 + mt * 16 + (lane >> 4) * 8;
#pragma unroll
    for (int i = 0; i < 8; ++i) {
      float v = acc[mt][i] + bv;
      if (res) v += res[(size_t)(r + i) * N + c];
      if (act) v = fmaxf(v, 0.0f);
      if (out) out[(size_t)(r + i) * N + c] = v;
      if (outh) outh[(size_t)(r + i) * N + c] = (_Float16)v;
    }
  }
}

// ---------------------------------------------------------------------------
// 32-row LDS-resident layer GEMM for PointNet: 2 row tiles per column group,
// one B fragment per K-step feeds 2 wmma. Output relu + bias to f16 LDS.
// ---------------------------------------------------------------------------
__device__ __forceinline__ void block_gemm32(const _Float16* Ain, int lda,
                                             const _Float16* __restrict__ Wp, int ktc,
                                             _Float16* Dout, const float* bias,
                                             int wave, int lane) {
#pragma unroll
  for (int tnt = 0; tnt < 2; ++tnt) {
    int c0 = (wave * 2 + tnt) * 16;
    v8f a0 = {}, a1 = {};
    for (int kt = 0; kt < ktc; ++kt) {
      v16h b = b_frag_p(Wp, ktc, kt, c0 >> 4, lane);
      a0 = wmma_f16(a_frag_h(Ain, lda, 0, kt << 5, lane), b, a0);
      a1 = wmma_f16(a_frag_h(Ain, lda, 16, kt << 5, lane), b, a1);
    }
    store_h(Dout, HD, 0, c0, a0, bias, true, lane);
    store_h(Dout, HD, 16, c0, a1, bias, true, lane);
  }
}

// ---------------------------------------------------------------------------
// PointNet / MLP3: per-token block. x[P,K0] -> 3x (GEMM+ReLU) -> masked max.
// ---------------------------------------------------------------------------
__global__ void pointnet_kernel(const float* __restrict__ attr, const int* __restrict__ valid,
                                const _Float16* __restrict__ W0p, const float* __restrict__ b0,
                                const _Float16* __restrict__ W1p, const float* __restrict__ b1,
                                const _Float16* __restrict__ W2p, const float* __restrict__ b2,
                                float* __restrict__ emb, _Float16* __restrict__ embh,
                                int* __restrict__ vr_i, float* __restrict__ vr_f,
                                int P, int K0, int K0pad) {
  __shared__ __align__(16) _Float16 xh[32 * 96];
  __shared__ __align__(16) _Float16 hA[32 * HD];
  __shared__ __align__(16) _Float16 hB[32 * HD];
  __shared__ __align__(16) _Float16 h3[32 * HD];
  __shared__ int vld[32];
  int tid = threadIdx.x, lane = tid & 31, wave = tid >> 5;
  int tok = blockIdx.x;

  for (int i = tid; i < 32 * 96; i += 256) xh[i] = (_Float16)0.0f;
  if (tid < 32) vld[tid] = (tid < P) ? valid[(size_t)tok * P + tid] : 0;
  __syncthreads();
  for (int i = tid; i < P * K0; i += 256) {
    int p = i / K0, k = i - p * K0;
    xh[p * 96 + k] = (_Float16)attr[((size_t)tok * P + p) * K0 + k];
  }
  __syncthreads();

  block_gemm32(xh, 96, W0p, K0pad >> 5, hA, b0, wave, lane);
  __syncthreads();
  block_gemm32(hA, HD, W1p, 8, hB, b1, wave, lane);
  __syncthreads();
  block_gemm32(hB, HD, W2p, 8, h3, b2, wave, lane);
  __syncthreads();

  // masked max over points (col = tid)
  float m = -1e30f;
  int any = 0;
  for (int p = 0; p < P; ++p)
    if (vld[p]) { any = 1; m = fmaxf(m, (float)h3[p * HD + tid]); }
  float val = any ? m : 0.0f;
  emb[(size_t)tok * HD + tid] = val;
  embh[(size_t)tok * HD + tid] = (_Float16)val;
  if (tid == 0) { vr_i[tok] = any; vr_f[tok] = any ? 1.0f : 0.0f; }
}

// ---------------------------------------------------------------------------
// KNN: one wave per token. Lane-cached keys + shfl_xor argmin reductions,
// deterministic lexicographic tie-breaking to match jax.lax.top_k.
// ---------------------------------------------------------------------------
__global__ void knn_kernel(const int* __restrict__ vr, const float* __restrict__ rel_dist,
                           const float* __restrict__ lim, int per_tok, int n, int off,
                           int* __restrict__ idx_o, int* __restrict__ inv_o) {
  int wv = blockIdx.x * (blockDim.x >> 5) + (threadIdx.x >> 5);
  if (wv >= NS * n) return;
  int lane = threadIdx.x & 31;
  int s = wv / n, i = wv - s * n;
  const float* drow = rel_dist + ((size_t)s * NTOT + off + i) * NTOT + off;
  const int* vrs = vr + (size_t)s * n;
  float limit = per_tok ? lim[wv] : lim[0];
  int cnt = n >> 5;
  float keys[32];
#pragma unroll
  for (int q = 0; q < 32; ++q) {
    if (q >= cnt) break;
    int j = (q << 5) + lane;
    float d = drow[j];
    keys[q] = (vrs[j] && d < limit) ? d : INFINITY;
  }
  float pk = -INFINITY;
  int pj = -1;
  for (int jj = 0; jj < KNN; ++jj) {
    float bk = INFINITY;
    int bj = 0x7fffffff;
#pragma unroll
    for (int q = 0; q < 32; ++q) {
      if (q >= cnt) break;
      int j = (q << 5) + lane;
      float k = keys[q];
      bool after = (k > pk) || (k == pk && j > pj);
      bool better = (k < bk) || (k == bk && j < bj);
      if (after && better) { bk = k; bj = j; }
    }
#pragma unroll
    for (int o = 16; o > 0; o >>= 1) {
      float ok_ = __shfl_xor(bk, o, 32);
      int oj = __shfl_xor(bj, o, 32);
      if (ok_ < bk || (ok_ == bk && oj < bj)) { bk = ok_; bj = oj; }
    }
    int invf = (bk == INFINITY) || (bj >= n);
    if (lane == 0) {
      idx_o[(size_t)wv * KNN + jj] = (bj >= n) ? 0 : bj;
      inv_o[(size_t)wv * KNN + jj] = invf;
    }
    pk = bk;
    pj = (bj >= n) ? n : bj;
  }
}

// ---------------------------------------------------------------------------
// 48-row LDS-resident stage GEMM for attention: 3 row tiles per column group,
// one B fragment per K-step feeds 3 wmma.
// ---------------------------------------------------------------------------
__device__ __forceinline__ void block_gemm48(const _Float16* Ain, int lda,
                                             const _Float16* __restrict__ Wp, int ktc,
                                             _Float16* Dout, const float* bias,
                                             bool relu, int wave, int lane) {
#pragma unroll
  for (int tnt = 0; tnt < 2; ++tnt) {
    int c0 = (wave * 2 + tnt) * 16;
    v8f a0 = {}, a1 = {}, a2 = {};
    for (int kt = 0; kt < ktc; ++kt) {
      v16h b = b_frag_p(Wp, ktc, kt, c0 >> 4, lane);
      a0 = wmma_f16(a_frag_h(Ain, lda, 0, kt << 5, lane), b, a0);
      a1 = wmma_f16(a_frag_h(Ain, lda, 16, kt << 5, lane), b, a1);
      a2 = wmma_f16(a_frag_h(Ain, lda, 32, kt << 5, lane), b, a2);
    }
    store_h(Dout, HD, 0, c0, a0, bias, relu, lane);
    store_h(Dout, HD, 16, c0, a1, bias, relu, lane);
    store_h(Dout, HD, 32, c0, a2, bias, relu, lane);
  }
}

__device__ __forceinline__ void gather_store(_Float16* Dout, const float* __restrict__ Gsrc,
                                             const int* sidx, size_t rowbase, int r0,
                                             int c0, v8f acc, int lane) {
  int r = r0 + (lane >> 4) * 8, c = c0 + (lane & 15);
#pragma unroll
  for (int i = 0; i < 8; ++i) {
    int row = r + i;
    float g = (row < KNN) ? Gsrc[(rowbase + sidx[row]) * HD + c] : 0.0f;
    Dout[row * HD + c] = (_Float16)(acc[i] + g);
  }
}

__device__ __forceinline__ void block_gemm48_gather(const _Float16* Ain,
                                                    const _Float16* __restrict__ Wp,
                                                    const float* __restrict__ Gsrc,
                                                    const int* sidx, size_t rowbase,
                                                    _Float16* Dout, int wave, int lane) {
#pragma unroll
  for (int tnt = 0; tnt < 2; ++tnt) {
    int c0 = (wave * 2 + tnt) * 16;
    v8f a0 = {}, a1 = {}, a2 = {};
    for (int kt = 0; kt < 8; ++kt) {
      v16h b = b_frag_p(Wp, 8, kt, c0 >> 4, lane);
      a0 = wmma_f16(a_frag_h(Ain, HD, 0, kt << 5, lane), b, a0);
      a1 = wmma_f16(a_frag_h(Ain, HD, 16, kt << 5, lane), b, a1);
      a2 = wmma_f16(a_frag_h(Ain, HD, 32, kt << 5, lane), b, a2);
    }
    gather_store(Dout, Gsrc, sidx, rowbase, 0, c0, a0, lane);
    gather_store(Dout, Gsrc, sidx, rowbase, 16, c0, a1, lane);
    gather_store(Dout, Gsrc, sidx, rowbase, 32, c0, a2, lane);
  }
}

// ---------------------------------------------------------------------------
// Fused per-token attention with RPE: one block per token.
// ---------------------------------------------------------------------------
__global__ void attn_kernel(const float* __restrict__ Qb, const float* __restrict__ Kb,
                            const float* __restrict__ Vb, const int* __restrict__ idx_knn,
                            const int* __restrict__ inv_knn, const float* __restrict__ rel_pose,
                            int off, int n,
                            const _Float16* __restrict__ rW1p, const float* __restrict__ rb1,
                            const _Float16* __restrict__ rW2p, const float* __restrict__ rb2,
                            const _Float16* __restrict__ Wrkp, const _Float16* __restrict__ Wrvp,
                            _Float16* __restrict__ Obufh) {
  __shared__ __align__(16) _Float16 R0[48 * HD];
  __shared__ __align__(16) _Float16 R1[48 * HD];
  __shared__ __align__(16) _Float16 fh[48 * 32];
  __shared__ int sidx[KNN];
  __shared__ int sinv[KNN];
  __shared__ float qv[HD];
  __shared__ float logit[NH * KNN];
  __shared__ float wts[NH * KNN];
  int tid = threadIdx.x, lane = tid & 31, wave = tid >> 5;
  int tok = blockIdx.x;
  int s = tok / n, i = tok - s * n;
  size_t rowbase = (size_t)s * n;

  if (tid < KNN) {
    sidx[tid] = idx_knn[(size_t)tok * KNN + tid];
    sinv[tid] = inv_knn[(size_t)tok * KNN + tid];
  }
  for (int x = tid; x < 48 * 32; x += 256) fh[x] = (_Float16)0.0f;
  qv[tid] = Qb[(size_t)tok * HD + tid];
  __syncthreads();
  if (tid < KNN) {
    int j = sidx[tid];
    const float* pp = rel_pose + (((size_t)s * NTOT + off + i) * NTOT + off + j) * 3;
    fh[tid * 32 + 0] = (_Float16)pp[0];
    fh[tid * 32 + 1] = (_Float16)pp[1];
    fh[tid * 32 + 2] = (_Float16)cosf(pp[2]);
    fh[tid * 32 + 3] = (_Float16)sinf(pp[2]);
  }
  __syncthreads();

  // h1 = relu(f @ rW1 + rb1) -> R0   (K padded 4->32 at pack time)
  block_gemm48(fh, 32, rW1p, 1, R0, rb1, true, wave, lane);
  __syncthreads();
  // rpe = h1 @ rW2 + rb2 -> R1
  block_gemm48(R0, HD, rW2p, 8, R1, rb2, false, wave, lane);
  __syncthreads();
  // kp = rpe @ Wrk + gather(Kb) -> R0
  block_gemm48_gather(R1, Wrkp, Kb, sidx, rowbase, R0, wave, lane);
  __syncthreads();
  // logits (scale = 1/sqrt(64))
  if (tid < NH * KNN) {
    int h = tid / KNN, j = tid - h * KNN;
    float a = 0.0f;
    for (int d = 0; d < HD / NH; ++d)
      a += qv[h * (HD / NH) + d] * (float)R0[j * HD + h * (HD / NH) + d];
    logit[tid] = sinv[j] ? NEGB : a * 0.125f;
  }
  __syncthreads();
  if (tid < NH) {
    float mx = -INFINITY;
    for (int j = 0; j < KNN; ++j) mx = fmaxf(mx, logit[tid * KNN + j]);
    float sum = 0.0f;
    for (int j = 0; j < KNN; ++j) {
      float e = __expf(logit[tid * KNN + j] - mx);
      wts[tid * KNN + j] = e;
      sum += e;
    }
    float is = 1.0f / sum;
    for (int j = 0; j < KNN; ++j) wts[tid * KNN + j] *= is;
  }
  __syncthreads();
  // vp = rpe @ Wrv + gather(Vb) -> R0
  block_gemm48_gather(R1, Wrvp, Vb, sidx, rowbase, R0, wave, lane);
  __syncthreads();
  // o[c] = sum_j wts[head(c)][j] * vp[j][c]
  {
    int h = tid >> 6;
    float o = 0.0f;
    for (int j = 0; j < KNN; ++j) o += wts[h * KNN + j] * (float)R0[j * HD + tid];
    Obufh[(size_t)tok * HD + tid] = (_Float16)o;
  }
}

// ---------------------------------------------------------------------------
// LayerNorm over HD dims; optional zeroing of invalid tokens; f32+f16 outputs
// ---------------------------------------------------------------------------
__global__ void ln_kernel(const float* __restrict__ x, const float* __restrict__ g,
                          const float* __restrict__ b, const int* __restrict__ vr,
                          float* __restrict__ out, _Float16* __restrict__ outh) {
  __shared__ float red[HD];
  int tid = threadIdx.x, tok = blockIdx.x;
  float v = x[(size_t)tok * HD + tid];
  red[tid] = v;
  __syncthreads();
  for (int st = HD / 2; st > 0; st >>= 1) {
    if (tid < st) red[tid] += red[tid + st];
    __syncthreads();
  }
  float mu = red[0] * (1.0f / HD);
  __syncthreads();
  float d = v - mu;
  red[tid] = d * d;
  __syncthreads();
  for (int st = HD / 2; st > 0; st >>= 1) {
    if (tid < st) red[tid] += red[tid + st];
    __syncthreads();
  }
  float var = red[0] * (1.0f / HD);
  float y = d * rsqrtf(var + 1e-5f) * g[tid] + b[tid];
  if (vr && !vr[tok]) y = 0.0f;
  out[(size_t)tok * HD + tid] = y;
  if (outh) outh[(size_t)tok * HD + tid] = (_Float16)y;
}

// ---------------------------------------------------------------------------
// Host orchestration
// ---------------------------------------------------------------------------
struct TFP {
  _Float16 *Wq[6], *Wk[6], *Wv[6], *Wo[6], *Wrk[6], *Wrv[6], *fW1[6], *fW2[6];
  _Float16 *rW1, *rW2;
  const float *bq, *bk, *bv, *bo, *rb1, *rb2;
  const float *ln1g, *ln1b, *fb1, *fb2, *ln2g, *ln2b;
};

static void run_branch(hipStream_t stream, int n, int off, int P, int K0, int K0pad,
                       const float* attr, const int* valid,
                       const _Float16* W0p, const float* b0,
                       const _Float16* W1p, const float* b1,
                       const _Float16* W2p, const float* b2,
                       const float* lim, int per_tok, int L,
                       const float* rel_pose, const float* rel_dist, const TFP& w,
                       float* emb, _Float16* embh, int* vr, int* idxb, int* invb,
                       float* Qb, float* Kb, float* Vb, _Float16* Obh,
                       float* xb, _Float16* xbh, float* tmp, _Float16* ffhh,
                       float* out_emb, float* out_vr) {
  int toks = NS * n;
  pointnet_kernel<<<toks, 256, 0, stream>>>(attr, valid, W0p, b0, W1p, b1, W2p, b2,
                                            emb, embh, vr, out_vr, P, K0, K0pad);
  knn_kernel<<<(toks + 3) / 4, 128, 0, stream>>>(vr, rel_dist, lim, per_tok, n, off,
                                                 idxb, invb);
  dim3 gh(toks / 64, HD / 128);
  dim3 gf(toks / 64, FFD / 128);
  for (int li = 0; li < 2; ++li) {
    int l = L + li;
    gemm_ws<<<gh, 256, 0, stream>>>(embh, w.Wq[l], w.bq + l * HD, nullptr,
                                    Qb, nullptr, HD, HD, 0);
    gemm_ws<<<gh, 256, 0, stream>>>(embh, w.Wk[l], w.bk + l * HD, nullptr,
                                    Kb, nullptr, HD, HD, 0);
    gemm_ws<<<gh, 256, 0, stream>>>(embh, w.Wv[l], w.bv + l * HD, nullptr,
                                    Vb, nullptr, HD, HD, 0);
    attn_kernel<<<toks, 256, 0, stream>>>(Qb, Kb, Vb, idxb, invb, rel_pose, off, n,
                                          w.rW1, w.rb1, w.rW2, w.rb2,
                                          w.Wrk[l], w.Wrv[l], Obh);
    gemm_ws<<<gh, 256, 0, stream>>>(Obh, w.Wo[l], w.bo + l * HD, emb,
                                    tmp, nullptr, HD, HD, 0);
    ln_kernel<<<toks, 256, 0, stream>>>(tmp, w.ln1g + l * HD, w.ln1b + l * HD, nullptr,
                                        xb, xbh);
    gemm_ws<<<gf, 256, 0, stream>>>(xbh, w.fW1[l], w.fb1 + l * FFD, nullptr,
                                    nullptr, ffhh, FFD, HD, 1);
    gemm_ws<<<gh, 256, 0, stream>>>(ffhh, w.fW2[l], w.fb2 + l * HD, xb,
                                    tmp, nullptr, HD, FFD, 0);
    ln_kernel<<<toks, 256, 0, stream>>>(tmp, w.ln2g + l * HD, w.ln2b + l * HD, vr,
                                        (li == 1) ? out_emb : emb, embh);
  }
}

extern "C" void kernel_launch(void* const* d_in, const int* in_sizes, int n_in,
                              void* d_out, int out_size, void* d_ws, size_t ws_size,
                              hipStream_t stream) {
  (void)in_sizes; (void)n_in; (void)out_size; (void)ws_size;
  const int*   agent_valid = (const int*)d_in[2];
  const float* agent_attr  = (const float*)d_in[3];
  const int*   map_valid   = (const int*)d_in[4];
  const float* map_attr    = (const float*)d_in[5];
  const int*   tl_valid    = (const int*)d_in[6];
  const float* tl_attr     = (const float*)d_in[7];
  const float* rel_pose    = (const float*)d_in[8];
  const float* rel_dist    = (const float*)d_in[9];
  const float* lim_map     = (const float*)d_in[10];
  const float* lim_tl      = (const float*)d_in[11];
  const float* lim_agent   = (const float*)d_in[12];

  char* p = (char*)d_ws;
  auto alloc = [&](size_t bytes) -> char* {
    char* r = p;
    p += (bytes + 255) & ~(size_t)255;
    return r;
  };
  auto pack = [&](const float* W, int N, int K, int Kpad) -> _Float16* {
    _Float16* dst = (_Float16*)alloc((size_t)N * Kpad * 2);
    int total = (N >> 4) * (Kpad >> 5) * 32;
    pack_b<<<(total + 255) / 256, 256, 0, stream>>>(W, dst, N, K, Kpad);
    return dst;
  };

  const int maxtok = NS * NM;
  float* emb      = (float*)alloc((size_t)maxtok * HD * 4);
  _Float16* embh  = (_Float16*)alloc((size_t)maxtok * HD * 2);
  float* Qb       = (float*)alloc((size_t)maxtok * HD * 4);
  float* Kb       = (float*)alloc((size_t)maxtok * HD * 4);
  float* Vb       = (float*)alloc((size_t)maxtok * HD * 4);
  _Float16* Obh   = (_Float16*)alloc((size_t)maxtok * HD * 2);
  float* xb       = (float*)alloc((size_t)maxtok * HD * 4);
  _Float16* xbh   = (_Float16*)alloc((size_t)maxtok * HD * 2);
  float* tmp      = (float*)alloc((size_t)maxtok * HD * 4);
  _Float16* ffhh  = (_Float16*)alloc((size_t)maxtok * FFD * 2);
  int* vr    = (int*)alloc((size_t)maxtok * 4);
  int* idxb  = (int*)alloc((size_t)maxtok * KNN * 4);
  int* invb  = (int*)alloc((size_t)maxtok * KNN * 4);

  // ---- pack all B-side weights to fragment-ordered f16 ----
  const float* pn_map_W0 = (const float*)d_in[13];
  const float* pn_map_b0 = (const float*)d_in[14];
  const float* pn_map_W  = (const float*)d_in[15];
  const float* pn_map_b  = (const float*)d_in[16];
  const float* pn_ag_W0  = (const float*)d_in[17];
  const float* pn_ag_b0  = (const float*)d_in[18];
  const float* pn_ag_W   = (const float*)d_in[19];
  const float* pn_ag_b   = (const float*)d_in[20];
  const float* tl_W0     = (const float*)d_in[21];
  const float* tl_b0     = (const float*)d_in[22];
  const float* tl_W      = (const float*)d_in[23];
  const float* tl_b      = (const float*)d_in[24];

  _Float16* mapW0p = pack(pn_map_W0, HD, DMA, 32);
  _Float16* mapW1p = pack(pn_map_W, HD, HD, HD);
  _Float16* mapW2p = pack(pn_map_W + (size_t)HD * HD, HD, HD, HD);
  _Float16* agW0p  = pack(pn_ag_W0, HD, DAA, 96);
  _Float16* agW1p  = pack(pn_ag_W, HD, HD, HD);
  _Float16* agW2p  = pack(pn_ag_W + (size_t)HD * HD, HD, HD, HD);
  _Float16* tlW0p  = pack(tl_W0, HD, DTLA, 32);
  _Float16* tlW1p  = pack(tl_W, HD, HD, HD);
  _Float16* tlW2p  = pack(tl_W + (size_t)HD * HD, HD, HD, HD);

  TFP w;
  w.rW1 = pack((const float*)d_in[25], HD, 4, 32);
  w.rb1 = (const float*)d_in[26];
  w.rW2 = pack((const float*)d_in[27], HD, HD, HD);
  w.rb2 = (const float*)d_in[28];
  const float* tf_Wq = (const float*)d_in[29];
  const float* tf_Wk = (const float*)d_in[31];
  const float* tf_Wv = (const float*)d_in[33];
  const float* tf_Wo = (const float*)d_in[35];
  const float* tf_Wrk = (const float*)d_in[37];
  const float* tf_Wrv = (const float*)d_in[38];
  const float* tf_fW1 = (const float*)d_in[41];
  const float* tf_fW2 = (const float*)d_in[43];
  for (int l = 0; l < 6; ++l) {
    w.Wq[l]  = pack(tf_Wq + (size_t)l * HD * HD, HD, HD, HD);
    w.Wk[l]  = pack(tf_Wk + (size_t)l * HD * HD, HD, HD, HD);
    w.Wv[l]  = pack(tf_Wv + (size_t)l * HD * HD, HD, HD, HD);
    w.Wo[l]  = pack(tf_Wo + (size_t)l * HD * HD, HD, HD, HD);
    w.Wrk[l] = pack(tf_Wrk + (size_t)l * HD * HD, HD, HD, HD);
    w.Wrv[l] = pack(tf_Wrv + (size_t)l * HD * HD, HD, HD, HD);
    w.fW1[l] = pack(tf_fW1 + (size_t)l * HD * FFD, FFD, HD, HD);
    w.fW2[l] = pack(tf_fW2 + (size_t)l * FFD * HD, HD, FFD, FFD);
  }
  w.bq = (const float*)d_in[30];  w.bk = (const float*)d_in[32];
  w.bv = (const float*)d_in[34];  w.bo = (const float*)d_in[36];
  w.ln1g = (const float*)d_in[39]; w.ln1b = (const float*)d_in[40];
  w.fb1 = (const float*)d_in[42];  w.fb2 = (const float*)d_in[44];
  w.ln2g = (const float*)d_in[45]; w.ln2b = (const float*)d_in[46];

  float* out = (float*)d_out;
  float* o_map_emb = out;
  float* o_map_vr  = o_map_emb + (size_t)NS * NM * HD;
  float* o_tl_emb  = o_map_vr + (size_t)NS * NM;
  float* o_tl_vr   = o_tl_emb + (size_t)NS * NTL * HD;
  float* o_ag_emb  = o_tl_vr + (size_t)NS * NTL;
  float* o_ag_vr   = o_ag_emb + (size_t)NS * NA * HD;

  // map branch (layers 0,1)
  run_branch(stream, NM, 0, NP, DMA, 32, map_attr, map_valid,
             mapW0p, pn_map_b0, mapW1p, pn_map_b, mapW2p, pn_map_b + HD,
             lim_map, 0, 0, rel_pose, rel_dist, w,
             emb, embh, vr, idxb, invb, Qb, Kb, Vb, Obh, xb, xbh, tmp, ffhh,
             o_map_emb, o_map_vr);
  // traffic-light branch (layers 2,3) — MLP3 as P=1 pointnet
  run_branch(stream, NTL, NM, 1, DTLA, 32, tl_attr, tl_valid,
             tlW0p, tl_b0, tlW1p, tl_b, tlW2p, tl_b + HD,
             lim_tl, 0, 2, rel_pose, rel_dist, w,
             emb, embh, vr, idxb, invb, Qb, Kb, Vb, Obh, xb, xbh, tmp, ffhh,
             o_tl_emb, o_tl_vr);
  // agent branch (layers 4,5)
  run_branch(stream, NA, NM + NTL, NTI, DAA, 96, agent_attr, agent_valid,
             agW0p, pn_ag_b0, agW1p, pn_ag_b, agW2p, pn_ag_b + HD,
             lim_agent, 1, 4, rel_pose, rel_dist, w,
             emb, embh, vr, idxb, invb, Qb, Kb, Vb, Obh, xb, xbh, tmp, ffhh,
             o_ag_emb, o_ag_vr);
}
